// SARANBlock_2869038153951
// MI455X (gfx1250) — matrix operand order repaired and checked by reading, not compiled, not run on hardware
//
#include <hip/hip_runtime.h>
#include <hip/hip_bf16.h>

typedef _Float16 half_t;
typedef half_t v16h __attribute__((ext_vector_type(16)));
typedef half_t v8h  __attribute__((ext_vector_type(8)));
typedef float  v8f  __attribute__((ext_vector_type(8)));
typedef unsigned int u32x4 __attribute__((ext_vector_type(4)));
typedef int i32x8 __attribute__((ext_vector_type(8)));
typedef int i32x4 __attribute__((ext_vector_type(4)));

#define N_EMBD 1024
#define SEQ    2048
#define NBATCH 8
#define MROWS  (NBATCH * SEQ)      // 16384
#define C3     (3 * N_EMBD)        // 3072

// ---------------------------------------------------------------------------
// Fragment loaders for 16-bit WMMA, contiguous k order (same in A and B).
// A (16x32, MxK): lane holds row M=lane%16, K halves kb..kb+15 contiguous,
//                 kb = (lane<16) ? 0 : 16.
// B (32x16, KxN): lane holds column N=lane%16, K halves kb..kb+15 contiguous,
//                 kb = (lane<16) ? 0 : 16.
// ---------------------------------------------------------------------------
__device__ __forceinline__ v16h load_frag_a(const half_t* base, int stride, int lane) {
    const half_t* p = base + (lane & 15) * stride + ((lane < 16) ? 0 : 16);
    union { v16h v; v8h h[2]; } u;
    u.h[0] = *(const v8h*)(p);
    u.h[1] = *(const v8h*)(p + 8);
    return u.v;
}

__device__ __forceinline__ v16h load_frag_b(const half_t* base, int stride, int lane) {
    const half_t* p = base + (lane & 15) * stride + ((lane < 16) ? 0 : 16);
    return *(const v16h*)(p);
}

// ---------------------------------------------------------------------------
// TDM: DMA a 64-row x 32-half 2D tile (row stride = strideElems halves) from
// global memory into LDS at byte offset lds_off.  D# per ISA ch.8:
//  group0: [1:0]=count=1 | [63:32]=lds_addr | [120:64]=global_addr | type=2
//  group1: data_size=2B, tensor dims huge (no OOB clamp), tile 32x64,
//          tensor_dim0_stride = strideElems.  groups 2/3 zero (2D tensor).
// Tracked by TENSORcnt; wave-level op (EXEC ignored) -> issue from one wave.
// This toolchain exposes the 6-operand clang-23 builtin:
//   (u32x4, i32x8, i32x4, i32x4, i32x8, i32 cpol)
// ---------------------------------------------------------------------------
__device__ __forceinline__ void tdm_load_tile_64x32(unsigned lds_off,
                                                    const half_t* g,
                                                    int strideElems) {
    unsigned long long ga = (unsigned long long)(uintptr_t)g;
    u32x4 g0;
    g0[0] = 1u;                                            // count=1, user mode
    g0[1] = lds_off;                                       // LDS byte address
    g0[2] = (unsigned)ga;                                  // global addr [31:0]
    g0[3] = (unsigned)((ga >> 32) & 0x01ffffffu)           // global addr [56:32]
          | 0x80000000u;                                   // type=2 (image)
    i32x8 g1;
    g1[0] = 0x00010000;                                    // data_size=1 (2 bytes)
    g1[1] = (int)0xffff0000u;                              // tensor_dim0 lo16
    g1[2] = (int)0xffff7fffu;                              // dim0 hi16 | dim1 lo16
    g1[3] = (int)(0x7fffu | (32u << 16));                  // dim1 hi16 | tile_dim0=32
    g1[4] = 64;                                            // tile_dim1=64, tile_dim2=0
    g1[5] = strideElems;                                   // dim0_stride [31:0]
    g1[6] = 0;                                             // dim0_stride hi, dim1_stride lo
    g1[7] = 0;
    i32x4 z4 = {0, 0, 0, 0};
    i32x8 z8 = {0, 0, 0, 0, 0, 0, 0, 0};
    __builtin_amdgcn_tensor_load_to_lds(g0, g1, z4, z4, z8, 0);
}

// ---------------------------------------------------------------------------
// f32 -> f16 cast
// ---------------------------------------------------------------------------
__global__ void cast_f16_kernel(const float* __restrict__ in, half_t* __restrict__ out, int n) {
    int i = blockIdx.x * blockDim.x + threadIdx.x;
    if (i < n) out[i] = (half_t)in[i];
}

// ---------------------------------------------------------------------------
// RMSNorm + cast to f16.  One 256-thread block per row of C=1024.
// ---------------------------------------------------------------------------
__global__ __launch_bounds__(256) void rmsnorm_cast_kernel(
    const float* __restrict__ x, const float* __restrict__ w,
    half_t* __restrict__ out) {
    const int row = blockIdx.x;
    const int tid = threadIdx.x;
    const float* xr = x + (size_t)row * N_EMBD;
    float4 v = ((const float4*)xr)[tid];
    float ss = v.x * v.x + v.y * v.y + v.z * v.z + v.w * v.w;
    #pragma unroll
    for (int m = 16; m >= 1; m >>= 1) ss += __shfl_xor(ss, m);
    __shared__ float wsum[8];
    if ((tid & 31) == 0) wsum[tid >> 5] = ss;
    __syncthreads();
    float tot = 0.f;
    #pragma unroll
    for (int i = 0; i < 8; ++i) tot += wsum[i];
    const float r = rsqrtf(tot * (1.0f / N_EMBD) + 1e-6f);
    float4 wv = ((const float4*)w)[tid];
    half_t* o = out + (size_t)row * N_EMBD + tid * 4;
    o[0] = (half_t)(v.x * r * wv.x);
    o[1] = (half_t)(v.y * r * wv.y);
    o[2] = (half_t)(v.z * r * wv.z);
    o[3] = (half_t)(v.w * r * wv.w);
}

// ---------------------------------------------------------------------------
// Generic WMMA GEMM: out[M,N] = act( A[M,K] * W[N,K]^T ) (+ resid[M,N]).
// 128 threads = 4 waves; 64x64 block tile, each wave a 32x32 quadrant.
// A/B tiles staged to LDS by the Tensor Data Mover, double-buffered so the
// DMA of tile i+1 overlaps WMMA on tile i.  Only wave 0 issues TDM ops
// (EXEC is ignored by TDM, so a scalar branch avoids duplicate DMAs).
// ---------------------------------------------------------------------------
template <bool SILU, bool OUTF16>
__global__ __launch_bounds__(128) void gemm_kernel(
    const half_t* __restrict__ A, const half_t* __restrict__ W,
    const float* __restrict__ resid, void* __restrict__ outp,
    int M, int N, int K) {
    __shared__ half_t As[2][64 * 32];
    __shared__ half_t Bs[2][64 * 32];

    const int tid  = threadIdx.x;
    const int lane = tid & 31;
    const int wv   = tid >> 5;
    const int mb   = blockIdx.y * 64;
    const int nb   = blockIdx.x * 64;
    const int wrow = (wv >> 1) * 32;
    const int wcol = (wv & 1) * 32;

    const bool issuer = (__builtin_amdgcn_readfirstlane(wv) == 0);
    const unsigned as_off = (unsigned)(uintptr_t)&As[0][0];
    const unsigned bs_off = (unsigned)(uintptr_t)&Bs[0][0];
    const half_t* aTile = A + (size_t)mb * K;
    const half_t* wTile = W + (size_t)nb * K;

    v8f acc[2][2] = {};

    const int nk = K >> 5;
    if (issuer) {
        tdm_load_tile_64x32(as_off, aTile, K);
        tdm_load_tile_64x32(bs_off, wTile, K);
    }

    int buf = 0;
    for (int i = 0; i < nk; ++i) {
        if (issuer) {
            if (i + 1 < nk) {
                const unsigned nxt = (unsigned)(buf ^ 1) * (unsigned)sizeof(As[0]);
                tdm_load_tile_64x32(as_off + nxt, aTile + (size_t)(i + 1) * 32, K);
                tdm_load_tile_64x32(bs_off + nxt, wTile + (size_t)(i + 1) * 32, K);
                __builtin_amdgcn_s_wait_tensorcnt(2);   // current buffer landed
            } else {
                __builtin_amdgcn_s_wait_tensorcnt(0);
            }
        }
        __syncthreads();                                // release buffer to all waves

        v16h af0 = load_frag_a(&As[buf][(wrow     ) * 32], 32, lane);
        v16h af1 = load_frag_a(&As[buf][(wrow + 16) * 32], 32, lane);
        v16h bf0 = load_frag_b(&Bs[buf][(wcol     ) * 32], 32, lane);
        v16h bf1 = load_frag_b(&Bs[buf][(wcol + 16) * 32], 32, lane);

        acc[0][0] = __builtin_amdgcn_wmma_f32_16x16x32_f16(false, af0, false, bf0, (short)0, acc[0][0], false, false);
        acc[0][1] = __builtin_amdgcn_wmma_f32_16x16x32_f16(false, af0, false, bf1, (short)0, acc[0][1], false, false);
        acc[1][0] = __builtin_amdgcn_wmma_f32_16x16x32_f16(false, af1, false, bf0, (short)0, acc[1][0], false, false);
        acc[1][1] = __builtin_amdgcn_wmma_f32_16x16x32_f16(false, af1, false, bf1, (short)0, acc[1][1], false, false);

        __syncthreads();                                // all waves done with buffer
        buf ^= 1;
    }

    const int rhi = (lane >> 4) * 8;
    const int cn  = lane & 15;
    #pragma unroll
    for (int i = 0; i < 2; ++i)
        #pragma unroll
        for (int j = 0; j < 2; ++j)
            #pragma unroll
            for (int r = 0; r < 8; ++r) {
                const int grow = mb + wrow + i * 16 + rhi + r;
                const int gcol = nb + wcol + j * 16 + cn;
                float v = acc[i][j][r];
                if (SILU) v = v / (1.0f + __expf(-v));
                if (resid) v += resid[(size_t)grow * N + gcol];
                if (OUTF16) ((half_t*)outp)[(size_t)grow * N + gcol] = (half_t)v;
                else        ((float*)outp)[(size_t)grow * N + gcol] = v;
            }
}

// ---------------------------------------------------------------------------
// Causal attention for one (batch, 16-query tile).  128 threads = 4 waves.
// Dynamic LDS: Qs 16x1024 f16 (32KB) | S 16x2048 f32 (128KB) | P 16x2048 f16 (64KB)
// qkv layout: [B*T][3072] f16 with q|k|v at column offsets 0|1024|2048.
// ---------------------------------------------------------------------------
#define ATTN_SMEM (16 * 1024 * 2 + 16 * 2048 * 4 + 16 * 2048 * 2)

__global__ __launch_bounds__(128) void attn_kernel(
    const half_t* __restrict__ qkv, half_t* __restrict__ av) {
    extern __shared__ char smem[];
    half_t* Qs = (half_t*)smem;                                  // 16 x 1024
    float*  S  = (float*)(smem + 16 * 1024 * 2);                 // 16 x 2048
    half_t* P  = (half_t*)(smem + 16 * 1024 * 2 + 16 * 2048 * 4);// 16 x 2048

    const int b     = blockIdx.x >> 7;
    const int qt    = blockIdx.x & 127;
    const int qbase = qt * 16;
    const int tid   = threadIdx.x;
    const int lane  = tid & 31;
    const int wv    = tid >> 5;

    // stage Q tile: 16 rows x 1024 halves
    for (int i = 0; i < 16; ++i) {
        const int idx = i * 128 + tid;          // 2048 chunks of 8 halves
        const int row = idx >> 7;
        const int cho = (idx & 127) * 8;
        *(uint4*)&Qs[row * 1024 + cho] =
            *(const uint4*)(qkv + ((size_t)(b * SEQ + qbase + row)) * C3 + cho);
    }
    __syncthreads();

    // ---- score pass: S[16, key] = (Q . K) * C^-0.5, causal-masked ----
    const int nkt = qt + 1;                     // key tiles of 16
    for (int kt = wv; kt < nkt; kt += 4) {
        v8f c = {};
        const int key = kt * 16 + (lane & 15);
        const half_t* krow = qkv + ((size_t)(b * SEQ + key)) * C3 + N_EMBD
                           + ((lane < 16) ? 0 : 16);
        for (int ks = 0; ks < N_EMBD; ks += 32) {
            v16h a  = load_frag_a(Qs + ks, 1024, lane);
            v16h bf = *(const v16h*)(krow + ks);
            c = __builtin_amdgcn_wmma_f32_16x16x32_f16(false, a, false, bf, (short)0, c, false, false);
        }
        #pragma unroll
        for (int r = 0; r < 8; ++r) {
            const int row  = r + ((lane >> 4) << 3);
            const int kcol = kt * 16 + (lane & 15);
            float v = c[r] * 0.03125f;          // 1024^-0.5
            if (kcol > qbase + row) v = -1e30f; // causal mask
            S[row * 2048 + kcol] = v;
        }
    }
    __syncthreads();

    // ---- softmax over each row (8 threads per row) ----
    const int keycount = qbase + 16;
    const int keypad   = (keycount + 31) & ~31;
    {
        const int row = tid >> 3;
        const int g   = tid & 7;
        float mx = -1e30f;
        for (int c = g; c < keycount; c += 8) mx = fmaxf(mx, S[row * 2048 + c]);
        #pragma unroll
        for (int m = 4; m >= 1; m >>= 1) mx = fmaxf(mx, __shfl_xor(mx, m));
        float sum = 0.f;
        for (int c = g; c < keycount; c += 8) sum += __expf(S[row * 2048 + c] - mx);
        #pragma unroll
        for (int m = 4; m >= 1; m >>= 1) sum += __shfl_xor(sum, m);
        const float inv = 1.0f / sum;
        for (int c = g; c < keycount; c += 8)
            P[row * 2048 + c] = (half_t)(__expf(S[row * 2048 + c] - mx) * inv);
        for (int c = keycount + g; c < keypad; c += 8)
            P[row * 2048 + c] = (half_t)0.f;
    }
    __syncthreads();

    // ---- PV pass: av[16, C] = P[16, keypad] * V[keypad, C] ----
    const int nkc = keypad >> 5;
    for (int nt = wv; nt < 64; nt += 4) {
        v8f c = {};
        const int col = nt * 16 + (lane & 15);
        const half_t* vbase = qkv + ((size_t)b * SEQ) * C3 + 2 * N_EMBD + col;
        const int kb = (lane < 16) ? 0 : 16;
        for (int kc = 0; kc < nkc; ++kc) {
            v16h a = load_frag_a(P + kc * 32, 2048, lane);
            v16h bf;
            #pragma unroll
            for (int j = 0; j < 16; ++j)
                bf[j] = vbase[(size_t)(kc * 32 + kb + j) * C3];
            c = __builtin_amdgcn_wmma_f32_16x16x32_f16(false, a, false, bf, (short)0, c, false, false);
        }
        #pragma unroll
        for (int r = 0; r < 8; ++r) {
            const int row = r + ((lane >> 4) << 3);
            av[((size_t)(b * SEQ + qbase + row)) * N_EMBD + col] = (half_t)c[r];
        }
    }
}

// ---------------------------------------------------------------------------
// Host side
// ---------------------------------------------------------------------------
extern "C" void kernel_launch(void* const* d_in, const int* in_sizes, int n_in,
                              void* d_out, int out_size, void* d_ws, size_t ws_size,
                              hipStream_t stream) {
    const float* x     = (const float*)d_in[0];
    const float* ln1_w = (const float*)d_in[1];
    const float* ln2_w = (const float*)d_in[2];
    const float* qkv_w = (const float*)d_in[3];
    const float* out_w = (const float*)d_in[4];
    const float* w1    = (const float*)d_in[5];
    const float* w2    = (const float*)d_in[6];

    char* ws = (char*)d_ws;
    const size_t MB = 1024ull * 1024ull;
    half_t* h   = (half_t*)(ws + 0);            // 32MB  [M, C]
    half_t* qkv = (half_t*)(ws + 32 * MB);      // 96MB  [M, 3C]
    half_t* av  = (half_t*)(ws + 128 * MB);     // 32MB  [M, C]
    float*  x1  = (float*) (ws + 160 * MB);     // 64MB  [M, C]
    half_t* h2  = (half_t*)(ws + 224 * MB);     // 32MB  [M, C]
    half_t* ff  = (half_t*)(ws + 256 * MB);     // 64MB  [M, 2C]
    half_t* wq  = (half_t*)(ws + 320 * MB);     // 6MB
    half_t* wo  = (half_t*)(ws + 326 * MB);     // 2MB
    half_t* w1h = (half_t*)(ws + 328 * MB);     // 4MB
    half_t* w2h = (half_t*)(ws + 332 * MB);     // 4MB

    // weight casts
    {
        int n;
        n = C3 * N_EMBD;
        cast_f16_kernel<<<(n + 255) / 256, 256, 0, stream>>>(qkv_w, wq, n);
        n = N_EMBD * N_EMBD;
        cast_f16_kernel<<<(n + 255) / 256, 256, 0, stream>>>(out_w, wo, n);
        n = 2 * N_EMBD * N_EMBD;
        cast_f16_kernel<<<(n + 255) / 256, 256, 0, stream>>>(w1, w1h, n);
        n = N_EMBD * 2 * N_EMBD;
        cast_f16_kernel<<<(n + 255) / 256, 256, 0, stream>>>(w2, w2h, n);
    }

    // h = rmsnorm(x, ln1_w) in f16
    rmsnorm_cast_kernel<<<MROWS, 256, 0, stream>>>(x, ln1_w, h);

    // qkv = h @ qkv_w^T  (f16 out)
    gemm_kernel<false, true><<<dim3(C3 / 64, MROWS / 64), 128, 0, stream>>>(
        h, wq, nullptr, qkv, MROWS, C3, N_EMBD);

    // attention -> av (f16)
    static bool attn_attr_set = []() {
        (void)hipFuncSetAttribute((const void*)attn_kernel,
                                  hipFuncAttributeMaxDynamicSharedMemorySize, ATTN_SMEM);
        return true;
    }();
    (void)attn_attr_set;
    attn_kernel<<<NBATCH * (SEQ / 16), 128, ATTN_SMEM, stream>>>(qkv, av);

    // x1 = x + av @ out_w^T  (f32 out)
    gemm_kernel<false, false><<<dim3(N_EMBD / 64, MROWS / 64), 128, 0, stream>>>(
        av, wo, x, (void*)x1, MROWS, N_EMBD, N_EMBD);

    // h2 = rmsnorm(x1, ln2_w) in f16
    rmsnorm_cast_kernel<<<MROWS, 256, 0, stream>>>(x1, ln2_w, h2);

    // ff = silu(h2 @ w1^T)  (f16 out)
    gemm_kernel<true, true><<<dim3(2 * N_EMBD / 64, MROWS / 64), 128, 0, stream>>>(
        h2, w1h, nullptr, ff, MROWS, 2 * N_EMBD, N_EMBD);

    // out = x1 + ff @ w2^T  (f32 out)
    gemm_kernel<false, false><<<dim3(N_EMBD / 64, MROWS / 64), 128, 0, stream>>>(
        ff, w2h, x1, d_out, MROWS, N_EMBD, 2 * N_EMBD);
}
